// MambaBackbone_21182778704402
// MI455X (gfx1250) — matrix operand order repaired and checked by reading, not compiled, not run on hardware
//
#include <hip/hip_runtime.h>
#include <hip/hip_bf16.h>
#include <math.h>

typedef __bf16 bf16;
typedef bf16  v16bf __attribute__((ext_vector_type(16)));
typedef float v8f   __attribute__((ext_vector_type(8)));

#define BATCH   512
#define DMODEL  768
#define DINNER  1536
#define DSTATE  16
#define DCONV   4
#define DTRANK  48
#define XDBN    80
#define DEPTH   24

// epilogue parameter block (unused fields null per mode)
struct Epi {
  const float* bias;                                    // EPI 0
  const float* cs; const float* cw; const float* cb;    // EPI 1 (conv)
  float*       xact_out;                                // EPI 1 output
  const float* dtb; const float* xact; const float* xdb;// EPI 2 (ssm)
  const float* alog; const float* Dpv; const float* ss;
  const float* zbuf;
};

// split one fp32 vector into bf16 hi/lo halves (f ~= hi + lo)
__device__ __forceinline__ void cvt16(const float* f, v16bf& hi, v16bf& lo) {
#pragma unroll
  for (int e = 0; e < 16; ++e) {
    const bf16 h = (bf16)f[e];
    hi[e] = h;
    lo[e] = (bf16)(f[e] - (float)h);
  }
}

__device__ __forceinline__ void load16(const float* p, float* f) {
  *(float4*)&f[0]  = *(const float4*)(p);
  *(float4*)&f[4]  = *(const float4*)(p + 4);
  *(float4*)&f[8]  = *(const float4*)(p + 8);
  *(float4*)&f[12] = *(const float4*)(p + 12);
}

__device__ __forceinline__ float sigmoidf_(float v) { return 1.f / (1.f + __expf(-v)); }

// ---------------------------------------------------------------------------
// C[M,N] = A[M,K] (leading dim LDA) * W[N,K]^T, fused epilogue.
// fp32 accuracy via 3x bf16 split WMMA (AhBh + AhBl + AlBh), fp32 accumulate.
// One wave: 32(M) x 64(N) tile; 2 A-subtiles x 4 B-subtiles -> 8 accumulators.
// EPI: 0 = (+bias) store   1 = conv+SiLU / compact-z split   2 = fused SSM
// Requires: M%32==0, N%16==0, K%16==0 (true for every call site).
// ---------------------------------------------------------------------------
template<int K, int LDA, int EPI>
__global__ __launch_bounds__(256) void wmma_gemm_nt(
    const float* __restrict__ A, const float* __restrict__ W,
    float* __restrict__ C, int M, int N, Epi ep)
{
  const int lane  = threadIdx.x & 31;
  const int nb    = (N + 63) >> 6;
  const int wid   = blockIdx.x * (blockDim.x >> 5) + (threadIdx.x >> 5);
  const int total = (M >> 5) * nb;
  if (wid >= total) return;                    // wave-uniform
  const int m0 = (wid / nb) << 5;              // 32 rows per wave
  const int n0 = (wid % nb) << 6;              // 64 cols per wave

  const int bcol = lane & 15;                  // also A row-in-subtile
  const int kgA  = (lane >> 4) << 3;           // A half-wave K offset: 0/8
  const int kgB  = (lane >> 4) << 3;           // B half-wave K offset: 0/8

  const float* ap0 = A + (size_t)(m0 + bcol) * LDA;
  const float* ap1 = A + (size_t)(m0 + 16 + bcol) * LDA;
  const float* wp0 = W + (size_t)(n0 + bcol) * K;   // subtile t adds t*16*K

  int nt_valid = (N - n0) >> 4;                // uniform per wave; N%16==0
  if (nt_valid > 4) nt_valid = 4;

  v8f acc[2][4] = {};
  constexpr int KFULL = (K / 32) * 32;

  for (int k0 = 0; k0 < KFULL; k0 += 32) {
    // prefetch next k-step of the streaming W operand
    if (k0 + 32 < KFULL) {
#pragma unroll
      for (int t = 0; t < 4; ++t)
        if (t < nt_valid)
          __builtin_prefetch(wp0 + (size_t)(t << 4) * K + k0 + 32 + kgB, 0, 3);
    }

    float af[16];
    v16bf ah[2], al[2];
    *(float4*)&af[0]  = *(const float4*)(ap0 + k0 + kgA);
    *(float4*)&af[4]  = *(const float4*)(ap0 + k0 + kgA + 4);
    *(float4*)&af[8]  = *(const float4*)(ap0 + k0 + 16 + kgA);
    *(float4*)&af[12] = *(const float4*)(ap0 + k0 + 16 + kgA + 4);
    cvt16(af, ah[0], al[0]);
    *(float4*)&af[0]  = *(const float4*)(ap1 + k0 + kgA);
    *(float4*)&af[4]  = *(const float4*)(ap1 + k0 + kgA + 4);
    *(float4*)&af[8]  = *(const float4*)(ap1 + k0 + 16 + kgA);
    *(float4*)&af[12] = *(const float4*)(ap1 + k0 + 16 + kgA + 4);
    cvt16(af, ah[1], al[1]);

#pragma unroll
    for (int t = 0; t < 4; ++t) {
      if (t >= nt_valid) break;                // uniform
      float bf_[16];
      {
        const float* bp = wp0 + (size_t)(t << 4) * K + k0 + kgB;
        *(float4*)&bf_[0]  = *(const float4*)(bp);
        *(float4*)&bf_[4]  = *(const float4*)(bp + 4);
        *(float4*)&bf_[8]  = *(const float4*)(bp + 16);
        *(float4*)&bf_[12] = *(const float4*)(bp + 20);
      }
      v16bf bh, bl; cvt16(bf_, bh, bl);
#pragma unroll
      for (int m = 0; m < 2; ++m) {
        acc[m][t] = __builtin_amdgcn_wmma_f32_16x16x32_bf16(false, ah[m], false, bh, (short)0, acc[m][t], false, false);
        acc[m][t] = __builtin_amdgcn_wmma_f32_16x16x32_bf16(false, ah[m], false, bl, (short)0, acc[m][t], false, false);
        acc[m][t] = __builtin_amdgcn_wmma_f32_16x16x32_bf16(false, al[m], false, bh, (short)0, acc[m][t], false, false);
      }
    }
  }

  if constexpr ((K % 32) == 16) {              // 16-wide K tail (dt-proj: K=48)
    constexpr int k0 = KFULL;
    float af[16];
    v16bf ah[2], al[2];
#pragma unroll
    for (int e = 8; e < 16; ++e) af[e] = 0.f;  // K..K+15 region is zero
    *(float4*)&af[0] = *(const float4*)(ap0 + k0 + kgA);
    *(float4*)&af[4] = *(const float4*)(ap0 + k0 + kgA + 4);
    cvt16(af, ah[0], al[0]);
    *(float4*)&af[0] = *(const float4*)(ap1 + k0 + kgA);
    *(float4*)&af[4] = *(const float4*)(ap1 + k0 + kgA + 4);
    cvt16(af, ah[1], al[1]);

#pragma unroll
    for (int t = 0; t < 4; ++t) {
      if (t >= nt_valid) break;
      float bf_[16];
      {
        const float* bp = wp0 + (size_t)(t << 4) * K + k0 + kgB;
        *(float4*)&bf_[0] = *(const float4*)(bp);
        *(float4*)&bf_[4] = *(const float4*)(bp + 4);
#pragma unroll
        for (int e = 8; e < 16; ++e) bf_[e] = 0.f;
      }
      v16bf bh, bl; cvt16(bf_, bh, bl);
#pragma unroll
      for (int m = 0; m < 2; ++m) {
        acc[m][t] = __builtin_amdgcn_wmma_f32_16x16x32_bf16(false, ah[m], false, bh, (short)0, acc[m][t], false, false);
        acc[m][t] = __builtin_amdgcn_wmma_f32_16x16x32_bf16(false, ah[m], false, bl, (short)0, acc[m][t], false, false);
        acc[m][t] = __builtin_amdgcn_wmma_f32_16x16x32_bf16(false, al[m], false, bh, (short)0, acc[m][t], false, false);
      }
    }
  }

  // -------- epilogue.  C/D layout: VGPR r -> rows base+r / base+8+r ---------
  const int rhalf = (lane >> 4) << 3;
#pragma unroll
  for (int t = 0; t < 4; ++t) {
    if (t >= nt_valid) break;                  // uniform
    const int col = n0 + (t << 4) + bcol;      // < N (N%16==0)

    if constexpr (EPI == 0) {                  // plain (+bias)
      const float bv = ep.bias ? ep.bias[col] : 0.f;
#pragma unroll
      for (int m = 0; m < 2; ++m) {
        const int rbase = m0 + (m << 4) + rhalf;
#pragma unroll
        for (int r = 0; r < 8; ++r)
          C[(size_t)(rbase + r) * N + col] = acc[m][t][r] + bv;
      }
    } else if constexpr (EPI == 1) {           // xz-GEMM: conv+SiLU / compact z
      if (col < DINNER) {                      // uniform per subtile
        const float4 w4 = *(const float4*)(ep.cw + (size_t)col * DCONV);
        const float cbv = ep.cb[col];
#pragma unroll
        for (int m = 0; m < 2; ++m) {
          const int rbase = m0 + (m << 4) + rhalf;
#pragma unroll
          for (int r = 0; r < 8; ++r) {
            const int row = rbase + r;
            const size_t idx = (size_t)row * DINNER + col;
            const float4 c4 = *(const float4*)(ep.cs + idx * DCONV);
            const float s = c4.y * w4.x + c4.z * w4.y + c4.w * w4.z
                          + acc[m][t][r] * w4.w + cbv;
            ep.xact_out[idx] = s * sigmoidf_(s);
          }
        }
      } else {                                 // z half -> compact buffer
#pragma unroll
        for (int m = 0; m < 2; ++m) {
          const int rbase = m0 + (m << 4) + rhalf;
#pragma unroll
          for (int r = 0; r < 8; ++r)
            C[(size_t)(rbase + r) * DINNER + (col - DINNER)] = acc[m][t][r];
        }
      }
    } else {                                   // EPI == 2: dt-GEMM + fused SSM
      float av[16];
      load16(ep.alog + (size_t)col * DSTATE, av);
      const float dtbv = ep.dtb[col];
      const float dv   = ep.Dpv[col];
#pragma unroll
      for (int m = 0; m < 2; ++m) {
        const int rbase = m0 + (m << 4) + rhalf;
#pragma unroll
        for (int r = 0; r < 8; ++r) {
          const int row = rbase + r;
          const size_t idx = (size_t)row * DINNER + col;
          float dt = acc[m][t][r] + dtbv;
          dt = (dt > 20.f) ? dt : log1pf(__expf(dt));         // softplus
          const float x   = ep.xact[idx];
          const float dtx = dt * x;
          const float* xr = ep.xdb + (size_t)row * XDBN;
          float sv[16], Bv[16], Cv[16];
          load16(ep.ss + idx * DSTATE, sv);
          load16(xr + DTRANK, Bv);
          load16(xr + DTRANK + DSTATE, Cv);
          float a2 = 0.f;
#pragma unroll
          for (int n = 0; n < DSTATE; ++n) {
            const float dA = __expf(-dt * __expf(av[n]));     // exp(dt*(-exp(A_log)))
            a2 += (sv[n] * dA + dtx * Bv[n]) * Cv[n];
          }
          float yv = a2 + dv * x;
          const float z = ep.zbuf[idx];
          yv *= z * sigmoidf_(z);                             // * silu(z)
          C[idx] = yv;
        }
      }
    }
  }
}

// ---------------------------------------------------------------------------
// final LayerNorm over rows of 768
// ---------------------------------------------------------------------------
__global__ __launch_bounds__(256) void ln_kernel(
    const float* __restrict__ h, const float* __restrict__ g,
    const float* __restrict__ bb, float* __restrict__ out)
{
  __shared__ float red0[8], red1[8];
  const int b = blockIdx.x;
  const float* row = h + (size_t)b * DMODEL;
  float s = 0.f, s2 = 0.f;
  for (int j = threadIdx.x; j < DMODEL; j += blockDim.x) {
    const float v = row[j]; s += v; s2 += v * v;
  }
  for (int off = 16; off; off >>= 1) {
    s  += __shfl_down(s, off, 32);
    s2 += __shfl_down(s2, off, 32);
  }
  const int w = threadIdx.x >> 5;
  if ((threadIdx.x & 31) == 0) { red0[w] = s; red1[w] = s2; }
  __syncthreads();
  if (threadIdx.x == 0) {
    float ts = 0.f, ts2 = 0.f;
    for (int i = 0; i < 8; ++i) { ts += red0[i]; ts2 += red1[i]; }
    red0[0] = ts; red1[0] = ts2;
  }
  __syncthreads();
  const float mu   = red0[0] * (1.f / DMODEL);
  const float var  = red1[0] * (1.f / DMODEL) - mu * mu;
  const float rstd = rsqrtf(var + 1e-5f);
  for (int j = threadIdx.x; j < DMODEL; j += blockDim.x)
    out[(size_t)b * DMODEL + j] = (row[j] - mu) * rstd * g[j] + bb[j];
}

// ---------------------------------------------------------------------------
static inline int gemm_blocks(int M, int N) {
  const int nb    = (N + 63) / 64;
  const int waves = (M / 32) * nb;
  return (waves + 7) / 8;                     // 8 waves per 256-thread block
}

extern "C" void kernel_launch(void* const* d_in, const int* in_sizes, int n_in,
                              void* d_out, int out_size, void* d_ws, size_t ws_size,
                              hipStream_t stream)
{
  const float* x_t        = (const float*)d_in[0];
  const float* in_w       = (const float*)d_in[1];
  const float* in_b       = (const float*)d_in[2];
  const float* xz_w       = (const float*)d_in[3];
  const float* conv_w     = (const float*)d_in[4];
  const float* conv_b     = (const float*)d_in[5];
  const float* xproj_w    = (const float*)d_in[6];
  const float* dtproj_w   = (const float*)d_in[7];
  const float* dtproj_b   = (const float*)d_in[8];
  const float* A_log      = (const float*)d_in[9];
  const float* Dp         = (const float*)d_in[10];
  const float* outproj_w  = (const float*)d_in[11];
  const float* conv_state = (const float*)d_in[12];
  const float* ssm_state  = (const float*)d_in[13];
  const float* ln_g       = (const float*)d_in[14];
  const float* ln_b       = (const float*)d_in[15];
  float* out = (float*)d_out;

  float* ws   = (float*)d_ws;
  float* h    = ws; ws += (size_t)BATCH * DMODEL;       // 512x768
  float* zbuf = ws; ws += (size_t)BATCH * DINNER;       // 512x1536 (compact z)
  float* xact = ws; ws += (size_t)BATCH * DINNER;       // 512x1536
  float* xdb  = ws; ws += (size_t)BATCH * XDBN;         // 512x80
  float* yb   = ws; ws += (size_t)BATCH * DINNER;       // 512x1536

  Epi e0{}; e0.bias = in_b;
  // h = x_t @ in_w^T + in_b
  wmma_gemm_nt<64, 64, 0><<<gemm_blocks(BATCH, DMODEL), 256, 0, stream>>>(
      x_t, in_w, h, BATCH, DMODEL, e0);

  for (int l = 0; l < DEPTH; ++l) {
    const float* xzw = xz_w       + (size_t)l * 2 * DINNER * DMODEL;
    const float* cw  = conv_w     + (size_t)l * DINNER * DCONV;
    const float* cb  = conv_b     + (size_t)l * DINNER;
    const float* xpw = xproj_w    + (size_t)l * XDBN * DINNER;
    const float* dpw = dtproj_w   + (size_t)l * DINNER * DTRANK;
    const float* dpb = dtproj_b   + (size_t)l * DINNER;
    const float* al  = A_log      + (size_t)l * DINNER * DSTATE;
    const float* dl  = Dp         + (size_t)l * DINNER;
    const float* opw = outproj_w  + (size_t)l * DMODEL * DINNER;
    const float* csl = conv_state + (size_t)l * BATCH * DINNER * DCONV;
    const float* ssl = ssm_state  + (size_t)l * BATCH * DINNER * DSTATE;

    // xz GEMM (512 x 3072, K=768) with fused conv+SiLU (x half) / compact z
    Epi e1{}; e1.cs = csl; e1.cw = cw; e1.cb = cb; e1.xact_out = xact;
    wmma_gemm_nt<DMODEL, DMODEL, 1><<<gemm_blocks(BATCH, 2 * DINNER), 256, 0, stream>>>(
        h, xzw, zbuf, BATCH, 2 * DINNER, e1);

    // xdb = xact @ xproj_w^T (512 x 80, K=1536)
    Epi e2{};
    wmma_gemm_nt<DINNER, DINNER, 0><<<gemm_blocks(BATCH, XDBN), 256, 0, stream>>>(
        xact, xpw, xdb, BATCH, XDBN, e2);

    // dt GEMM (512 x 1536, K=48, lda=80) with fused softplus/SSM/gate -> yb
    Epi e3{}; e3.dtb = dpb; e3.xact = xact; e3.xdb = xdb; e3.alog = al;
    e3.Dpv = dl; e3.ss = ssl; e3.zbuf = zbuf;
    wmma_gemm_nt<DTRANK, XDBN, 2><<<gemm_blocks(BATCH, DINNER), 256, 0, stream>>>(
        xdb, dpw, yb, BATCH, DINNER, e3);

    // h = yb @ outproj_w^T (512 x 768, K=1536)
    Epi e4{};
    wmma_gemm_nt<DINNER, DINNER, 0><<<gemm_blocks(BATCH, DMODEL), 256, 0, stream>>>(
        yb, opw, h, BATCH, DMODEL, e4);
  }

  ln_kernel<<<BATCH, 256, 0, stream>>>(h, ln_g, ln_b, out);
}
